// DiscriminativeLoss_6433861009917
// MI455X (gfx1250) — matrix operand order repaired and checked by reading, not compiled, not run on hardware
//
#include <hip/hip_runtime.h>
#include <hip/hip_bf16.h>

// ---------------------------------------------------------------------------
// Problem constants (match reference setup_inputs)
// ---------------------------------------------------------------------------
constexpr int BB   = 4;            // batch
constexpr int DD   = 16;           // feature dims
constexpr int HH   = 512;
constexpr int WW   = 1024;
constexpr int NN   = HH * WW;      // 524288 points per image
constexpr int KC   = 24;           // clusters
constexpr int KP   = 32;           // padded clusters (two 16-col WMMA tiles)
constexpr float DELTA_VAR  = 1.0f;
constexpr float DELTA_DIST = 2.0f;

constexpr int NWAVES = 8;          // 256 threads/block = 8 wave32
constexpr int GX     = 64;         // segsum blocks per image
constexpr int NBLK   = BB * GX;    // total segsum blocks
constexpr int TILE   = KP * DD + KP;  // per-block partial: 512 sums + 32 counts

typedef __attribute__((ext_vector_type(16))) _Float16 v16h;
typedef __attribute__((ext_vector_type(8)))  float    v8f;

// Workspace layout (floats):
//   [0, NBLK*TILE)      per-block partial tiles (sums + counts)
//   then counts [B][KP], hv [B][KP], centers [B][KP][D]
constexpr int WS_PART    = 0;
constexpr int WS_COUNTS  = NBLK * TILE;                 // 139264
constexpr int WS_HV      = WS_COUNTS + BB * KP;         // +128
constexpr int WS_CENTERS = WS_HV + BB * KP;             // +128
constexpr int WS_TOTAL   = WS_CENTERS + BB * KP * DD;   // +2048

// ---------------------------------------------------------------------------
__global__ void dl_zero_kernel(float* p, int n) {
    int i = blockIdx.x * blockDim.x + threadIdx.x;
    if (i < n) p[i] = 0.0f;
}

// ---------------------------------------------------------------------------
// Pass 1: per-cluster sums via WMMA, transposed so no cross-lane gathers:
//   D[d][k] = sum_p Xt[d][p] * onehot[p][k]
// A (16x32 f16) = Xt tile: lane L holds dim L&15; K-slots are points
//   {0..7,16..23} (lanes 0-15) / {8..15,24..31} (lanes 16-31)
//   -> four 16B-aligned b128 loads per lane, cachelines fully consumed.
// B (32x16 f16) = one-hot: lane L holds cluster col L&15, K-slots = same points as A.
//   Second tile only needs clusters 16..23 (labels < 24) -> cols 24..31 stay zero.
// C (16x16 f32): lane L holds dim r+(L>=16?8:0), cluster col L&15.
// Per-block: 8 waves reduce into an LDS tile (ds_add_f32), then plain stores
// of one partial tile per block -> zero global atomics.
// ---------------------------------------------------------------------------
__global__ void dl_segsum_wmma_kernel(const float* __restrict__ data,
                                      const int*   __restrict__ labels,
                                      float* __restrict__ partials) {
    const int b    = blockIdx.y;
    const int lane = threadIdx.x & 31;

    __shared__ float        stile[KP * DD];
    __shared__ unsigned int hist[KP];
    for (int i = threadIdx.x; i < KP * DD; i += blockDim.x) stile[i] = 0.0f;
    for (int i = threadIdx.x; i < KP;      i += blockDim.x) hist[i]  = 0u;
    __syncthreads();

    const float* dptr = data   + (size_t)b * DD * NN;
    const int*   lptr = labels + (size_t)b * NN;

    const int   mrow    = lane & 15;               // dim handled by this lane (A row)
    const int   colbase = (lane < 16) ? 0 : 8;     // first point offset of this half
    const float* rowbase = dptr + (size_t)mrow * NN + colbase;

    v8f c1 = {};   // cluster columns 0..15
    v8f c2 = {};   // cluster columns 16..31 (24..31 stay zero)

    const int chunks      = NN / 32;                           // 16384 per image
    const int wave_global = blockIdx.x * NWAVES + (threadIdx.x >> 5);
    const int wave_stride = gridDim.x * NWAVES;

    for (int ch = wave_global; ch < chunks; ch += wave_stride) {
        const int p0 = ch * 32;

        // one label per lane: coalesced b32
        const int lab = lptr[p0 + lane];
        atomicAdd(&hist[lab], 1u);                 // ds_add (counts)

        // A tile: 16 f32 from this lane's dim-row, two runs of 8 points
        const float* r = rowbase + p0;
        float4 q0 = *(const float4*)(r);           // points +0..3
        float4 q1 = *(const float4*)(r + 4);       // points +4..7
        float4 q2 = *(const float4*)(r + 16);      // points +16..19
        float4 q3 = *(const float4*)(r + 20);      // points +20..23

        // B one-hot: lane holds cluster col mrow (+16), elements = same points as A.
        v16h b1, b2;                               // cols 24..31 stay zero (labels < 24)
        const int* lr = lptr + p0 + colbase;
#pragma unroll
        for (int e = 0; e < 16; ++e) {
            const int le = lr[(e < 8) ? e : e + 8];
            b1[e] = (le == mrow) ? (_Float16)1.0f : (_Float16)0.0f;
            b2[e] = (le == (mrow + 16)) ? (_Float16)1.0f : (_Float16)0.0f;
        }

        v16h a;
        a[0]  = (_Float16)q0.x;  a[1]  = (_Float16)q0.y;
        a[2]  = (_Float16)q0.z;  a[3]  = (_Float16)q0.w;
        a[4]  = (_Float16)q1.x;  a[5]  = (_Float16)q1.y;
        a[6]  = (_Float16)q1.z;  a[7]  = (_Float16)q1.w;
        a[8]  = (_Float16)q2.x;  a[9]  = (_Float16)q2.y;
        a[10] = (_Float16)q2.z;  a[11] = (_Float16)q2.w;
        a[12] = (_Float16)q3.x;  a[13] = (_Float16)q3.y;
        a[14] = (_Float16)q3.z;  a[15] = (_Float16)q3.w;

        c1 = __builtin_amdgcn_wmma_f32_16x16x32_f16(false, a, false, b1,
                                                    (short)0, c1, false, false);
        c2 = __builtin_amdgcn_wmma_f32_16x16x32_f16(false, a, false, b2,
                                                    (short)0, c2, false, false);
    }

    // Reduce the 8 waves' C tiles into the block LDS tile.
    // Lane L, VGPR r -> dim = r + (L>=16 ? 8 : 0), cluster col = L&15.
    const int cl   = lane & 15;
    const int doff = (lane >> 4) << 3;  // 0 or 8
#pragma unroll
    for (int r = 0; r < 8; ++r) {
        const int dim = r + doff;
        atomicAdd(&stile[cl * DD + dim],        c1[r]);   // ds_add_f32
        atomicAdd(&stile[(cl + 16) * DD + dim], c2[r]);
    }
    __syncthreads();

    // Plain stores of this block's partial tile (sums + counts).
    float* out = partials + (size_t)(b * GX + blockIdx.x) * TILE;
    for (int i = threadIdx.x; i < KP * DD; i += blockDim.x) out[i] = stile[i];
    for (int i = threadIdx.x; i < KP; i += blockDim.x)
        out[KP * DD + i] = (float)hist[i];
}

// ---------------------------------------------------------------------------
// Reduce per-block partials -> centers, counts  (2048 threads, trivial)
// ---------------------------------------------------------------------------
__global__ void dl_reduce_kernel(const float* __restrict__ partials,
                                 float* __restrict__ centers,
                                 float* __restrict__ counts) {
    int idx = blockIdx.x * blockDim.x + threadIdx.x;
    if (idx >= BB * KP * DD) return;
    int d = idx % DD;
    int k = (idx / DD) % KP;
    int b = idx / (KP * DD);

    const float* base = partials + (size_t)b * GX * TILE;
    float s = 0.0f, cnt = 0.0f;
    for (int j = 0; j < GX; ++j) {
        const float* t = base + (size_t)j * TILE;
        s   += t[k * DD + d];
        cnt += t[KP * DD + k];
    }
    centers[idx] = (k < KC && cnt > 0.0f) ? s / cnt : 0.0f;
    if (d == 0) counts[b * KP + k] = cnt;
}

// ---------------------------------------------------------------------------
// Pass 2: variance term. Centers cached in LDS and fetched as 4x b128 per
// point; hinge^2 accumulated per cluster with ds_add_f32, one global atomic
// per (block, cluster).
// ---------------------------------------------------------------------------
__global__ void dl_var_kernel(const float* __restrict__ data,
                              const int*   __restrict__ labels,
                              const float* __restrict__ centers,
                              float* __restrict__ hv) {
    const int b = blockIdx.y;
    __shared__ __align__(16) float cs[KC * DD];
    __shared__ float lhv[KC];

    for (int i = threadIdx.x; i < KC * DD; i += blockDim.x) {
        int k = i / DD, d = i % DD;
        cs[i] = centers[((size_t)b * KP + k) * DD + d];
    }
    for (int i = threadIdx.x; i < KC; i += blockDim.x) lhv[i] = 0.0f;
    __syncthreads();

    const float* dptr = data   + (size_t)b * DD * NN;
    const int*   lptr = labels + (size_t)b * NN;

    for (int p = blockIdx.x * blockDim.x + threadIdx.x; p < NN;
         p += gridDim.x * blockDim.x) {
        const int lab = lptr[p];
        const float4* c4 = (const float4*)&cs[lab * DD];  // 64B-aligned row
        float4 ca = c4[0], cb = c4[1], cc = c4[2], cd = c4[3];  // 4x ds_load_b128

        float d2 = 0.0f, t;
        t = dptr[(size_t)0  * NN + p] - ca.x; d2 = fmaf(t, t, d2);
        t = dptr[(size_t)1  * NN + p] - ca.y; d2 = fmaf(t, t, d2);
        t = dptr[(size_t)2  * NN + p] - ca.z; d2 = fmaf(t, t, d2);
        t = dptr[(size_t)3  * NN + p] - ca.w; d2 = fmaf(t, t, d2);
        t = dptr[(size_t)4  * NN + p] - cb.x; d2 = fmaf(t, t, d2);
        t = dptr[(size_t)5  * NN + p] - cb.y; d2 = fmaf(t, t, d2);
        t = dptr[(size_t)6  * NN + p] - cb.z; d2 = fmaf(t, t, d2);
        t = dptr[(size_t)7  * NN + p] - cb.w; d2 = fmaf(t, t, d2);
        t = dptr[(size_t)8  * NN + p] - cc.x; d2 = fmaf(t, t, d2);
        t = dptr[(size_t)9  * NN + p] - cc.y; d2 = fmaf(t, t, d2);
        t = dptr[(size_t)10 * NN + p] - cc.z; d2 = fmaf(t, t, d2);
        t = dptr[(size_t)11 * NN + p] - cc.w; d2 = fmaf(t, t, d2);
        t = dptr[(size_t)12 * NN + p] - cd.x; d2 = fmaf(t, t, d2);
        t = dptr[(size_t)13 * NN + p] - cd.y; d2 = fmaf(t, t, d2);
        t = dptr[(size_t)14 * NN + p] - cd.z; d2 = fmaf(t, t, d2);
        t = dptr[(size_t)15 * NN + p] - cd.w; d2 = fmaf(t, t, d2);

        float h = fmaxf(sqrtf(d2) - DELTA_VAR, 0.0f);
        atomicAdd(&lhv[lab], h * h);
    }

    __syncthreads();
    for (int i = threadIdx.x; i < KC; i += blockDim.x)
        atomicAdd(&hv[b * KP + i], lhv[i]);
}

// ---------------------------------------------------------------------------
// Final scalar: var + dist + reg, one block.
// ---------------------------------------------------------------------------
__global__ void dl_final_kernel(const float* __restrict__ centers,
                                const float* __restrict__ counts,
                                const float* __restrict__ hv,
                                float* __restrict__ out) {
    __shared__ float red[256];
    float acc = 0.0f;

    // variance term: sum_b,k hv/count
    for (int i = threadIdx.x; i < BB * KC; i += blockDim.x) {
        int b = i / KC, k = i % KC;
        float cnt = counts[b * KP + k];
        if (cnt > 0.0f) acc += hv[b * KP + k] / cnt;
    }

    // regularization: sum_b mean_k ||c||
    for (int i = threadIdx.x; i < BB * KC; i += blockDim.x) {
        int b = i / KC, k = i % KC;
        float s = 0.0f;
#pragma unroll
        for (int d = 0; d < DD; ++d) {
            float v = centers[((size_t)b * KP + k) * DD + d];
            s = fmaf(v, v, s);
        }
        acc += sqrtf(s) * (1.0f / KC);
    }

    // distance term (diagonal kept, as in reference: contributes delta^2 each)
    float dacc = 0.0f;
    for (int i = threadIdx.x; i < BB * KC * KC; i += blockDim.x) {
        int b = i / (KC * KC);
        int r = i % (KC * KC);
        int ki = r / KC, kj = r % KC;
        float cd = 0.0f;
        if (ki != kj) {
            float s = 0.0f;
#pragma unroll
            for (int d = 0; d < DD; ++d) {
                float diff = centers[((size_t)b * KP + ki) * DD + d]
                           - centers[((size_t)b * KP + kj) * DD + d];
                s = fmaf(diff, diff, s);
            }
            cd = sqrtf(s);
        }
        float hd = fmaxf(DELTA_DIST - cd, 0.0f);
        dacc += hd * hd;
    }
    acc += dacc * (1.0f / (2.0f * KC * (KC - 1)));

    red[threadIdx.x] = acc;
    __syncthreads();
    for (int off = blockDim.x / 2; off > 0; off >>= 1) {
        if ((int)threadIdx.x < off) red[threadIdx.x] += red[threadIdx.x + off];
        __syncthreads();
    }
    if (threadIdx.x == 0) out[0] = red[0] * (1.0f / BB);
}

// ---------------------------------------------------------------------------
extern "C" void kernel_launch(void* const* d_in, const int* in_sizes, int n_in,
                              void* d_out, int out_size, void* d_ws, size_t ws_size,
                              hipStream_t stream) {
    const float* data   = (const float*)d_in[0];
    const int*   labels = (const int*)d_in[1];
    float* out = (float*)d_out;
    float* ws  = (float*)d_ws;

    float* partials = ws + WS_PART;
    float* counts   = ws + WS_COUNTS;
    float* hv       = ws + WS_HV;
    float* centers  = ws + WS_CENTERS;

    // 1) zero the small accumulators (counts + hv)
    dl_zero_kernel<<<1, 256, 0, stream>>>(counts, 2 * BB * KP);

    // 2) WMMA segment sums + counts -> per-block partial tiles (no atomics)
    dim3 g1(GX, BB);
    dl_segsum_wmma_kernel<<<g1, 256, 0, stream>>>(data, labels, partials);

    // 3) reduce partials -> centers, counts
    dl_reduce_kernel<<<(BB * KP * DD + 255) / 256, 256, 0, stream>>>(partials, centers, counts);

    // 4) variance pass
    dim3 g3(256, BB);
    dl_var_kernel<<<g3, 256, 0, stream>>>(data, labels, centers, hv);

    // 5) final scalar
    dl_final_kernel<<<1, 256, 0, stream>>>(centers, counts, hv, out);
}
